// ResidualQuantizer_74423193305696
// MI455X (gfx1250) — matrix operand order repaired and checked by reading, not compiled, not run on hardware
//
#include <hip/hip_runtime.h>

// ---------------------------------------------------------------------------
// Residual VQ (NQ=8, K=1024, D=256, B=32768) for gfx1250 (MI455X, wave32).
// Score GEMM: v_wmma_f32_16x16x32_bf16. Codebook B tiles are shared by 4
// waves per workgroup and streamed with global_load_async_to_lds_b128
// (ASYNCcnt / s_wait_asynccnt) using double buffering. Exact f32 elsewhere.
// ---------------------------------------------------------------------------

#define NQ 8
#define KC 1024
#define DD 256
#define NB 32768
#define LSTRIDE 261 /* 256+5: row stride 5 mod 64 -> conflict-free frag reads */

#define WAVES 4
#define ROWS_PER_WG (WAVES * 16)          // 64
#define CHUNK_TILES 4                     // N-tiles (16 cols each) per chunk
#define NCHUNK (64 / CHUNK_TILES)         // 16 chunks of 64 columns
#define CHUNK_BYTES (CHUNK_TILES * 8 * 32 * 32)  // 32 KB
#define CHUNK_US (CHUNK_BYTES / 2)
#define ASYNC_PER_THREAD (CHUNK_BYTES / (WAVES * 32 * 16))  // 16 b128 / thread

typedef __attribute__((ext_vector_type(16))) __bf16 v16bf;
typedef __attribute__((ext_vector_type(16))) unsigned short v16us;
typedef __attribute__((ext_vector_type(8))) float v8f;

static __device__ __forceinline__ unsigned short f2bf(float f) {
  unsigned int u = __builtin_bit_cast(unsigned int, f);
  u += 0x7fffu + ((u >> 16) & 1u); // round-to-nearest-even
  return (unsigned short)(u >> 16);
}

// ---------------------------------------------------------------------------
// Prep 1: swizzle codebooks (f32 [s][k][d]) into per-lane bf16 B-fragments.
// Layout: [s][t(64 n-tiles)][f(8 k-frags)][lane(32)][16 bf16]
// B matrix 32x16 (KxN): lane<16 -> n=lane,    K = f*32 + 0..15
//                       lane>=16 -> n=lane-16, K = f*32 + 16..31
__global__ void prep_cb(const float* __restrict__ cb, unsigned short* __restrict__ cbh) {
  int gid = blockIdx.x * blockDim.x + threadIdx.x; // 131072 threads
  if (gid >= NQ * 64 * 8 * 32) return;
  int lane = gid & 31;
  int f = (gid >> 5) & 7;
  int t = (gid >> 8) & 63;
  int s = gid >> 14;
  int n = t * 16 + (lane & 15);
  int koff = (lane < 16) ? 0 : 16;
  const float* src = cb + ((size_t)s * KC + n) * DD + f * 32 + koff;
  unsigned short* dst = cbh + (size_t)gid * 16;
#pragma unroll
  for (int j = 0; j < 16; ++j) dst[j] = f2bf(src[j]);
}

// Prep 2: e2[s][k] = sum_d cb[s][k][d]^2
__global__ void prep_e2(const float* __restrict__ cb, float* __restrict__ e2) {
  int gid = blockIdx.x * blockDim.x + threadIdx.x;
  if (gid >= NQ * KC) return;
  const float* row = cb + (size_t)gid * DD;
  float acc = 0.f;
#pragma unroll 4
  for (int d = 0; d < DD; ++d) acc += row[d] * row[d];
  e2[gid] = acc;
}

__global__ void init_loss(float* loss) {
  if (threadIdx.x == 0) *loss = 0.f;
}

// Issue one 32KB chunk of B-fragments: 128 threads x 16 async b128 each.
static __device__ __forceinline__ void issue_chunk_async(
    const unsigned short* __restrict__ gsrc, unsigned lds_base, int tid) {
  unsigned long long g = (unsigned long long)(size_t)gsrc + (unsigned)tid * 16u;
  unsigned l = lds_base + (unsigned)tid * 16u;
#pragma unroll
  for (int r = 0; r < ASYNC_PER_THREAD; ++r) {
    asm volatile("global_load_async_to_lds_b128 %0, %1, off"
                 :: "v"(l), "v"(g) : "memory");
    l += WAVES * 32 * 16; // 2048B per round across 128 threads
    g += WAVES * 32 * 16;
  }
}

// ---------------------------------------------------------------------------
// Main: 4 waves / WG, each wave owns 16 rows; B tiles shared via async LDS.
// ---------------------------------------------------------------------------
__global__ __launch_bounds__(WAVES * 32) void rvq_main(
    const float* __restrict__ z_e, const float* __restrict__ cb,
    const unsigned short* __restrict__ cbh, const float* __restrict__ e2,
    float* __restrict__ out_zq, int* __restrict__ out_codes,
    float* __restrict__ out_loss) {
  __shared__ float res[WAVES][16][LSTRIDE];
  __shared__ __align__(32) unsigned short Bbuf[2][CHUNK_US];
  __shared__ int codesLDS[ROWS_PER_WG];

  const int tid = threadIdx.x;
  const int lane = tid & 31;
  const int wave = tid >> 5;
  const size_t rowbase = (size_t)blockIdx.x * ROWS_PER_WG + wave * 16;
  const int m_row = lane & 15;           // A row (and C/e2 column slot)
  const int koffA = (lane < 16) ? 0 : 16; // A-matrix half-lane K offset

  const unsigned bbase0 = (unsigned)(size_t)&Bbuf[0][0];
  const unsigned bbase1 = (unsigned)(size_t)&Bbuf[1][0];

  // Load this wave's z_e rows into LDS (coalesced along D)
  for (int m = 0; m < 16; ++m)
#pragma unroll
    for (int i = 0; i < 8; ++i) {
      int d = lane + 32 * i;
      res[wave][m][d] = z_e[(rowbase + m) * DD + d];
    }

  float loss_acc = 0.f;

  for (int s = 0; s < NQ; ++s) {
    __syncthreads(); // residual settled for this stage

    // --- Build bf16 A fragments (contiguous k = f*32 + 16h + s, same as B) ----------------------
    v16us a_us[8];
#pragma unroll
    for (int f = 0; f < 8; ++f) {
      const float* rp = &res[wave][m_row][f * 32 + koffA];
#pragma unroll
      for (int j = 0; j < 8; ++j) {
        a_us[f][j] = f2bf(rp[j]);
        a_us[f][8 + j] = f2bf(rp[8 + j]);
      }
    }

    float best[8];
    int bidx[8];
#pragma unroll
    for (int r = 0; r < 8; ++r) { best[r] = 3.4e38f; bidx[r] = 0x7fffffff; }

    const unsigned short* cbh_s = cbh + (size_t)s * (64 * 8 * 32 * 16);
    const float* e2s = e2 + s * KC;

    // --- Async double-buffered sweep over 1024 codes ----------------------
    issue_chunk_async(cbh_s, bbase0, tid);
    issue_chunk_async(cbh_s + CHUNK_US, bbase1, tid);

    for (int c = 0; c < NCHUNK; ++c) {
      if (c == NCHUNK - 1)
        asm volatile("s_wait_asynccnt 0" ::: "memory");
      else
        asm volatile("s_wait_asynccnt 16" ::: "memory"); // chunk c landed
      __syncthreads();                                   // visible to all waves

      const unsigned short* bufp = (c & 1) ? &Bbuf[1][0] : &Bbuf[0][0];
#pragma unroll
      for (int tt = 0; tt < CHUNK_TILES; ++tt) {
        v8f cacc = {};
#pragma unroll
        for (int f = 0; f < 8; ++f) {
          v16bf a = __builtin_bit_cast(v16bf, a_us[f]);
          v16bf b = *(const v16bf*)&bufp[((tt * 8 + f) * 32 + lane) * 16];
          cacc = __builtin_amdgcn_wmma_f32_16x16x32_bf16(
              false, a, false, b, (short)0, cacc, false, false);
        }
        const int n = (c * CHUNK_TILES + tt) * 16 + m_row;
        const float e2v = e2s[n];
#pragma unroll
        for (int r = 0; r < 8; ++r) {
          float sc = __builtin_fmaf(-2.0f, cacc[r], e2v);
          if (sc < best[r]) { best[r] = sc; bidx[r] = n; }
        }
      }
      __syncthreads(); // all waves done reading this buffer
      if (c + 2 < NCHUNK)
        issue_chunk_async(cbh_s + (size_t)(c + 2) * CHUNK_US,
                          (c & 1) ? bbase1 : bbase0, tid);
    }

    // --- Reduce (min, idx) across 16 lanes of each half-wave --------------
#pragma unroll
    for (int off = 1; off < 16; off <<= 1) {
#pragma unroll
      for (int r = 0; r < 8; ++r) {
        float ob = __shfl_xor(best[r], off, 32);
        int oi = __shfl_xor(bidx[r], off, 32);
        if (ob < best[r] || (ob == best[r] && oi < bidx[r])) {
          best[r] = ob; bidx[r] = oi;
        }
      }
    }
    if (lane == 0 || lane == 16) { // half 0 -> rows 0-7, half 1 -> rows 8-15
      int mb = (lane >> 4) * 8;
#pragma unroll
      for (int r = 0; r < 8; ++r) {
        codesLDS[wave * 16 + mb + r] = bidx[r];
        out_codes[(rowbase + mb + r) * NQ + s] = bidx[r];
      }
    }
    __syncthreads(); // codes visible

    // --- Residual update (f32) + loss: (z_q - res)^2 == new_res^2 --------
    for (int m = 0; m < 16; ++m) {
      const float* crow = cb + ((size_t)s * KC + codesLDS[wave * 16 + m]) * DD;
#pragma unroll
      for (int i = 0; i < 8; ++i) {
        int d = lane + 32 * i;
        float v = res[wave][m][d] - crow[d];
        res[wave][m][d] = v;
        loss_acc += v * v;
      }
    }
  }

  __syncthreads();
  // z_q_sum = z_e - final residual (telescoping identity)
  for (int m = 0; m < 16; ++m)
#pragma unroll
    for (int i = 0; i < 8; ++i) {
      int d = lane + 32 * i;
      out_zq[(rowbase + m) * DD + d] = z_e[(rowbase + m) * DD + d] - res[wave][m][d];
    }

  // loss = 1.25/(B*D) * sum over stages of ||new residual||^2
#pragma unroll
  for (int off = 16; off >= 1; off >>= 1)
    loss_acc += __shfl_xor(loss_acc, off, 32);
  if (lane == 0)
    atomicAdd(out_loss, loss_acc * (1.25f / ((float)NB * (float)DD)));
}

// ---------------------------------------------------------------------------
extern "C" void kernel_launch(void* const* d_in, const int* in_sizes, int n_in,
                              void* d_out, int out_size, void* d_ws, size_t ws_size,
                              hipStream_t stream) {
  const float* z_e = (const float*)d_in[0]; // (32768, 256) f32
  const float* cb = (const float*)d_in[1];  // (8, 1024, 256) f32

  // workspace: bf16 swizzled codebooks (4 MB) then e2 (32 KB)
  unsigned short* cbh = (unsigned short*)d_ws;
  float* e2 = (float*)((char*)d_ws + (size_t)NQ * KC * DD * sizeof(unsigned short));

  float* out = (float*)d_out;
  float* out_zq = out;                                        // 32768*256 f32
  int* out_codes = (int*)(out + (size_t)NB * DD);             // 32768*8 i32
  float* out_loss = out + (size_t)NB * DD + (size_t)NB * NQ;  // 1 f32

  prep_cb<<<512, 256, 0, stream>>>(cb, cbh);
  prep_e2<<<(NQ * KC + 255) / 256, 256, 0, stream>>>(cb, e2);
  init_loss<<<1, 32, 0, stream>>>(out_loss);
  rvq_main<<<NB / ROWS_PER_WG, WAVES * 32, 0, stream>>>(
      z_e, cb, cbh, e2, out_zq, out_codes, out_loss);
}
